// GraphAutoencoder_59528246722864
// MI455X (gfx1250) — matrix operand order repaired and checked by reading, not compiled, not run on hardware
//
#include <hip/hip_runtime.h>

#define N_NODES 10000
#define N_EDGES 320000
#define IN_DIM  512
#define HID     128

typedef __attribute__((ext_vector_type(16))) _Float16 v16h;
typedef __attribute__((ext_vector_type(8)))  _Float16 v8h;
typedef __attribute__((ext_vector_type(8)))  float    v8f;

// ---- WMMA fragment loaders (CDNA5 16-bit layouts, wave32) ----
// A (16x32, row-major source, row stride = K):
//   lanes 0-15  : row = lane,    v0..3 = K[kb..kb+7],   v4..7 = K[kb+16..kb+23]
//   lanes 16-31 : row = lane-16, v0..3 = K[kb+8..kb+15], v4..7 = K[kb+24..kb+31]
static __device__ inline v16h load_fragA(const _Float16* row, int kb, int hi8) {
  v8h lo = *(const v8h*)(row + kb + hi8);
  v8h hi = *(const v8h*)(row + kb + hi8 + 16);
  return __builtin_shufflevector(lo, hi, 0,1,2,3,4,5,6,7,8,9,10,11,12,13,14,15);
}
// B (32x16, fed from W^T stored [N,K] row-major):
//   lanes 0-15  : col = lane,    v0..3 = K[kb..kb+7],   v4..7 = K[kb+16..kb+23]
//   lanes 16-31 : col = lane-16, v0..3 = K[kb+8..kb+15], v4..7 = K[kb+24..kb+31]
static __device__ inline v16h load_fragB(const _Float16* row, int kb, int hi16) {
  v8h lo = *(const v8h*)(row + kb + hi16 / 2);
  v8h hi = *(const v8h*)(row + kb + hi16 / 2 + 16);
  return __builtin_shufflevector(lo, hi, 0,1,2,3,4,5,6,7,8,9,10,11,12,13,14,15);
}

#define WMMA_F16(a,b,c) __builtin_amdgcn_wmma_f32_16x16x32_f16(false,(a),false,(b),(short)0,(c),false,false)

// ---------------- graph bookkeeping kernels ----------------
__global__ void k_init_deg(float* deg) {
  int i = blockIdx.x * blockDim.x + threadIdx.x;
  if (i < N_NODES) deg[i] = 1.0f;                 // self-loop
}
__global__ void k_count_deg(const long long* dst, float* deg) {
  int e = blockIdx.x * blockDim.x + threadIdx.x;
  if (e < N_EDGES) atomicAdd(&deg[(int)dst[e]], 1.0f);
}
__global__ void k_dinv(const float* deg, float* dinv) {
  int i = blockIdx.x * blockDim.x + threadIdx.x;
  if (i < N_NODES) { float d = deg[i]; dinv[i] = d > 0.0f ? rsqrtf(d) : 0.0f; }
}
__global__ void k_cvt_f16(const float* s, _Float16* d, int n) {
  int i = blockIdx.x * blockDim.x + threadIdx.x;
  if (i < n) d[i] = (_Float16)s[i];
}
// dst[c*R + r] = src[r*C + c]   (f32 -> f16 transpose)
__global__ void k_transpose_cvt(const float* s, _Float16* d, int R, int C) {
  int i = blockIdx.x * blockDim.x + threadIdx.x;
  if (i < R * C) { int r = i / C, c = i - r * C; d[(size_t)c * R + r] = (_Float16)s[i]; }
}

// ---------------- encoder GEMM: h = x @ W_enc  (f16 WMMA, f32 acc) ----------------
// grid = 625 blocks (one M-tile each), 8 waves = 8 N-tiles of HID=128
__global__ void __launch_bounds__(256) k_enc_gemm(const _Float16* __restrict__ xh,
                                                  const _Float16* __restrict__ wT,   // [HID, IN_DIM]
                                                  float* __restrict__ h) {
  const int lane = threadIdx.x & 31;
  const int nt   = threadIdx.x >> 5;          // 0..7
  const int mt   = blockIdx.x;                // 0..624
  const int col  = lane & 15;
  const int hi8  = (lane >= 16) ? 8  : 0;
  const int hi16 = (lane >= 16) ? 16 : 0;
  const _Float16* arow = xh + (size_t)(mt * 16 + col) * IN_DIM;
  const _Float16* brow = wT + (size_t)(nt * 16 + col) * IN_DIM;
  v8f c = {};
  #pragma unroll 4
  for (int kb = 0; kb < IN_DIM; kb += 32) {
    v16h a = load_fragA(arow, kb, hi8);
    v16h b = load_fragB(brow, kb, hi16);
    c = WMMA_F16(a, b, c);
  }
  float* orow = h + (size_t)(mt * 16 + hi8) * HID + nt * 16 + col;
  #pragma unroll
  for (int vr = 0; vr < 8; ++vr) orow[(size_t)vr * HID] = c[vr];
}

// agg = self-loop term + bias: agg[i][c] = h[i][c]*dinv[i]^2 + b_enc[c]
__global__ void k_agg_init(const float* __restrict__ h, const float* __restrict__ dinv,
                           const float* __restrict__ b_enc, float* __restrict__ agg) {
  int idx = blockIdx.x * blockDim.x + threadIdx.x;
  if (idx < N_NODES * HID) {
    int i = idx >> 7, c = idx & 127;
    float di = dinv[i];
    agg[idx] = h[idx] * di * di + b_enc[c];
  }
}

// scatter-add messages: one wave covers one edge's 128 channels (4 per lane)
__global__ void k_edge_scatter(const long long* __restrict__ src, const long long* __restrict__ dst,
                               const float* __restrict__ h, const float* __restrict__ dinv,
                               float* __restrict__ agg) {
  int idx = blockIdx.x * blockDim.x + threadIdx.x;
  if (idx >= N_EDGES * 32) return;
  int e = idx >> 5, g = idx & 31;
  int s = (int)src[e], d = (int)dst[e];
  float w = dinv[s] * dinv[d];
  const float4 hv = *(const float4*)(h + (size_t)s * HID + g * 4);
  float* ap = agg + (size_t)d * HID + g * 4;
  atomicAdd(ap + 0, hv.x * w);
  atomicAdd(ap + 1, hv.y * w);
  atomicAdd(ap + 2, hv.z * w);
  atomicAdd(ap + 3, hv.w * w);
}

// ---------------- decoder GEMM + sigmoid ----------------
// Each wave: 2 M-tiles x 4 N-tiles (32x64 output), K=128 -> 32 v_wmma per wave.
// A/B both live in L2 (2.5 MB each); kernel is bound by the 400 MB output store.
__global__ void __launch_bounds__(256) k_dec_gemm(const _Float16* __restrict__ ench,   // [N,128]
                                                  const _Float16* __restrict__ wT,     // [N,128]
                                                  const float* __restrict__ b_dec,
                                                  float* __restrict__ out) {
  const int gw   = (blockIdx.x * blockDim.x + threadIdx.x) >> 5;
  const int lane = threadIdx.x & 31;
  const int MG = 313, NG = 157;               // ceil(625/2), ceil(625/4)
  const int mg = gw % MG, ng = gw / MG;
  if (ng >= NG) return;                        // wave-uniform
  const int col  = lane & 15;
  const int hi8  = (lane >= 16) ? 8  : 0;
  const int hi16 = (lane >= 16) ? 16 : 0;
  const int m0 = mg * 32;
  const bool has_m1 = (m0 + 16) < N_NODES;     // 625 M-tiles; last group has 1

  const _Float16* arow0 = ench + (size_t)(m0 + col) * HID;
  const _Float16* arow1 = arow0 + (size_t)16 * HID;
  v16h a0[4], a1[4];
  #pragma unroll
  for (int k = 0; k < 4; ++k) a0[k] = load_fragA(arow0, k * 32, hi8);
  if (has_m1) {
    #pragma unroll
    for (int k = 0; k < 4; ++k) a1[k] = load_fragA(arow1, k * 32, hi8);
  }

  #pragma unroll
  for (int t = 0; t < 4; ++t) {
    int ntile = ng * 4 + t;
    if (ntile >= 625) continue;                // wave-uniform
    int nb = ntile * 16;
    const _Float16* brow = wT + (size_t)(nb + col) * HID;
    v8f c0 = {}, c1 = {};
    #pragma unroll
    for (int k = 0; k < 4; ++k) {
      v16h b = load_fragB(brow, k * 32, hi16);
      c0 = WMMA_F16(a0[k], b, c0);
      if (has_m1) c1 = WMMA_F16(a1[k], b, c1);
    }
    float bias = b_dec[nb + col];
    float* o0 = out + (size_t)(m0 + hi8) * N_NODES + nb + col;
    #pragma unroll
    for (int vr = 0; vr < 8; ++vr) {
      float v = c0[vr] + bias;
      o0[(size_t)vr * N_NODES] = 1.0f / (1.0f + __expf(-v));
    }
    if (has_m1) {
      float* o1 = out + (size_t)(m0 + 16 + hi8) * N_NODES + nb + col;
      #pragma unroll
      for (int vr = 0; vr < 8; ++vr) {
        float v = c1[vr] + bias;
        o1[(size_t)vr * N_NODES] = 1.0f / (1.0f + __expf(-v));
      }
    }
  }
}

extern "C" void kernel_launch(void* const* d_in, const int* in_sizes, int n_in,
                              void* d_out, int out_size, void* d_ws, size_t ws_size,
                              hipStream_t stream) {
  const float*     x      = (const float*)d_in[0];
  const long long* ei     = (const long long*)d_in[1];   // [2, E] int64
  const float*     W_enc  = (const float*)d_in[2];       // [512,128]
  const float*     b_enc  = (const float*)d_in[3];
  const float*     W_dec  = (const float*)d_in[4];       // [128,10000]
  const float*     b_dec  = (const float*)d_in[5];
  float*           outp   = (float*)d_out;
  const long long* esrc = ei;
  const long long* edst = ei + N_EDGES;

  // workspace carve-out (256B aligned)
  char* base = (char*)d_ws;
  size_t off = 0;
  auto alloc = [&](size_t bytes) { char* p = base + off; off = (off + bytes + 255) & ~(size_t)255; return p; };
  float*    deg   = (float*)   alloc((size_t)N_NODES * 4);
  float*    dinv  = (float*)   alloc((size_t)N_NODES * 4);
  float*    h     = (float*)   alloc((size_t)N_NODES * HID * 4);
  float*    agg   = (float*)   alloc((size_t)N_NODES * HID * 4);
  _Float16* xh    = (_Float16*)alloc((size_t)N_NODES * IN_DIM * 2);
  _Float16* WencT = (_Float16*)alloc((size_t)HID * IN_DIM * 2);
  _Float16* WdecT = (_Float16*)alloc((size_t)N_NODES * HID * 2);
  _Float16* ench  = (_Float16*)alloc((size_t)N_NODES * HID * 2);
  (void)ws_size; (void)in_sizes; (void)n_in; (void)out_size;

  const int T = 256;
  k_init_deg  <<<(N_NODES + T - 1) / T, T, 0, stream>>>(deg);
  k_count_deg <<<(N_EDGES + T - 1) / T, T, 0, stream>>>(edst, deg);
  k_dinv      <<<(N_NODES + T - 1) / T, T, 0, stream>>>(deg, dinv);

  k_cvt_f16       <<<(N_NODES * IN_DIM + T - 1) / T, T, 0, stream>>>(x, xh, N_NODES * IN_DIM);
  k_transpose_cvt <<<(IN_DIM * HID + T - 1) / T, T, 0, stream>>>(W_enc, WencT, IN_DIM, HID);
  k_transpose_cvt <<<(HID * N_NODES + T - 1) / T, T, 0, stream>>>(W_dec, WdecT, HID, N_NODES);

  k_enc_gemm <<<N_NODES / 16, T, 0, stream>>>(xh, WencT, h);           // 625 blocks

  k_agg_init     <<<(N_NODES * HID + T - 1) / T, T, 0, stream>>>(h, dinv, b_enc, agg);
  k_edge_scatter <<<(N_EDGES * 32 + T - 1) / T, T, 0, stream>>>(esrc, edst, h, dinv, agg);

  k_cvt_f16 <<<(N_NODES * HID + T - 1) / T, T, 0, stream>>>(agg, ench, N_NODES * HID);

  const int MG = 313, NG = 157;
  int dec_blocks = (MG * NG + 7) / 8;                                   // 8 waves / block
  k_dec_gemm <<<dec_blocks, T, 0, stream>>>(ench, WdecT, b_dec, outp);
}
